// AxialAttention_15436112462099
// MI455X (gfx1250) — compile-verified, matrix operand order checked by reading, not run on hardware
//
#include <hip/hip_runtime.h>
#include <hip/hip_bf16.h>
#include <math.h>

// ---------------------------------------------------------------------------
// Types / helpers
// ---------------------------------------------------------------------------
typedef __attribute__((ext_vector_type(16))) __bf16 v16bf;
typedef __attribute__((ext_vector_type(8)))  float  v8f;
typedef __attribute__((ext_vector_type(4)))  int    v4i;

union AFrag { v16bf v; unsigned short u[16]; uint4 q[2]; };

__device__ __forceinline__ unsigned short f2bf(float f) {
  unsigned int u = __float_as_uint(f);
  u += 0x7FFFu + ((u >> 16) & 1u);            // round-to-nearest-even
  return (unsigned short)(u >> 16);
}
__device__ __forceinline__ float bf2f(unsigned short h) {
  return __uint_as_float(((unsigned int)h) << 16);
}

// ---------------------------------------------------------------------------
// Async global->LDS staging (CDNA5 GLOBAL_LOAD_ASYNC_TO_LDS_B128, ASYNCcnt).
// Builtin signature (from hipcc diagnostic): arg0 = v4i in addrspace(1),
// arg1 = v4i in addrspace(3), then imm offset + imm cpol.
// Falls back to a b128 register round-trip when the builtins are absent.
// ---------------------------------------------------------------------------
#if defined(__has_builtin)
#if __has_builtin(__builtin_amdgcn_global_load_async_to_lds_b128) && \
    __has_builtin(__builtin_amdgcn_s_wait_asynccnt)
#define USE_ASYNC_LDS 1
#endif
#endif
#ifndef USE_ASYNC_LDS
#define USE_ASYNC_LDS 0
#endif

__device__ __forceinline__ void async_copy_b128(unsigned short* lds_dst,
                                                const unsigned short* gsrc) {
#if USE_ASYNC_LDS
  __builtin_amdgcn_global_load_async_to_lds_b128(
      (__attribute__((address_space(1))) v4i*)gsrc,
      (__attribute__((address_space(3))) v4i*)lds_dst, 0, 0);
#else
  *(uint4*)lds_dst = *(const uint4*)gsrc;
#endif
}
__device__ __forceinline__ void async_wait() {
#if USE_ASYNC_LDS
  __builtin_amdgcn_s_wait_asynccnt(0);
#endif
}

// Unified 16x32 bf16 fragment load (ISA 7.12.2): lane&15 = row, (lane>>4)*8 =
// K base; a lane's 16 elements are two contiguous 8-element runs
// (k=base..base+7 -> vgprs 0..3, k=base+16..base+23 -> vgprs 4..7), so each
// fragment is exactly two b128 loads.  Works for A tiles (rows=M) and B^T
// tiles (rows=N, contiguous K), LDS or global.
__device__ __forceinline__ void load_frag(const unsigned short* __restrict__ p,
                                          int row0, int ld, int k0, AFrag& f) {
  int lane = threadIdx.x & 31;
  const unsigned short* r =
      p + (size_t)(row0 + (lane & 15)) * ld + k0 + ((lane >> 4) << 3);
  f.q[0] = *(const uint4*)r;
  f.q[1] = *(const uint4*)(r + 16);
}

// ---------------------------------------------------------------------------
// f32 -> bf16 converter (8 elements / thread, packed 16B stores)
// ---------------------------------------------------------------------------
__global__ void cvt_kernel(const float* __restrict__ src,
                           unsigned short* __restrict__ dst, long n) {
  long i = ((long)blockIdx.x * blockDim.x + threadIdx.x) * 8;
  if (i >= n) return;
  float4 a = *(const float4*)(src + i);
  float4 b = *(const float4*)(src + i + 4);
  uint4 o;
  o.x = (unsigned)f2bf(a.x) | ((unsigned)f2bf(a.y) << 16);
  o.y = (unsigned)f2bf(a.z) | ((unsigned)f2bf(a.w) << 16);
  o.z = (unsigned)f2bf(b.x) | ((unsigned)f2bf(b.y) << 16);
  o.w = (unsigned)f2bf(b.z) | ((unsigned)f2bf(b.w) << 16);
  *(uint4*)(dst + i) = o;
}

// f32 [K][N] -> bf16 transposed [N][K] (weights; one-time gather)
__global__ void cvt_t_kernel(const float* __restrict__ src,
                             unsigned short* __restrict__ dst, int K, int N) {
  long idx = (long)blockIdx.x * blockDim.x + threadIdx.x;
  if (idx >= (long)K * N) return;
  int k = (int)(idx / N), n = (int)(idx % N);
  dst[(size_t)n * K + k] = f2bf(src[idx]);
}

// ---------------------------------------------------------------------------
// bf16 GEMM: C[M,N] = A[M,K] * Bt[N,K]^T (+bias), bf16 out, f32 acc.
// Block tile 128x64, 8 waves of 32x32 (2x2 WMMA tiles), K-step 32.
// LDS tiles staged via async global->LDS; fragments are two b128 loads.
// ---------------------------------------------------------------------------
__global__ __launch_bounds__(256) void gemm_bf16_kernel(
    const unsigned short* __restrict__ A, const unsigned short* __restrict__ Bt,
    unsigned short* __restrict__ C, const float* __restrict__ bias,
    int M, int N, int K)
{
  __shared__ __align__(16) unsigned short As[128 * 32];
  __shared__ __align__(16) unsigned short Bs[64 * 32];
  int bm = blockIdx.y * 128, bn = blockIdx.x * 64;
  int wave = threadIdx.x >> 5;
  int wm = (wave >> 1) * 32, wn = (wave & 1) * 32;
  v8f acc[2][2];
#pragma unroll
  for (int i = 0; i < 2; ++i)
#pragma unroll
    for (int j = 0; j < 2; ++j) acc[i][j] = (v8f)0.0f;

  int arow = threadIdx.x >> 1, acol = (threadIdx.x & 1) * 16;
  int brow = threadIdx.x >> 2, bcol = (threadIdx.x & 3) * 8;
  for (int k0 = 0; k0 < K; k0 += 32) {
    const unsigned short* ag = A + (size_t)(bm + arow) * K + k0 + acol;
    async_copy_b128(&As[arow * 32 + acol], ag);
    async_copy_b128(&As[arow * 32 + acol + 8], ag + 8);
    async_copy_b128(&Bs[brow * 32 + bcol],
                    Bt + (size_t)(bn + brow) * K + k0 + bcol);
    if (k0 + 32 < K)
      __builtin_prefetch(A + (size_t)(bm + arow) * K + k0 + 32 + acol, 0, 1);
    async_wait();
    __syncthreads();
    AFrag af, bf;
#pragma unroll
    for (int i = 0; i < 2; ++i) {
      load_frag(As, wm + 16 * i, 32, 0, af);
#pragma unroll
      for (int j = 0; j < 2; ++j) {
        load_frag(Bs, wn + 16 * j, 32, 0, bf);
        acc[i][j] = __builtin_amdgcn_wmma_f32_16x16x32_bf16(
            false, af.v, false, bf.v, (short)0, acc[i][j], false, false);
      }
    }
    __syncthreads();
  }
  int lane = threadIdx.x & 31;
  int nin = lane & 15, mh = (lane >> 4) * 8;
#pragma unroll
  for (int i = 0; i < 2; ++i)
#pragma unroll
    for (int j = 0; j < 2; ++j)
#pragma unroll
      for (int r = 0; r < 8; ++r) {
        int m = bm + wm + 16 * i + mh + r;
        int n = bn + wn + 16 * j + nin;
        float v = acc[i][j][r];
        if (bias) v += bias[n];
        C[(size_t)m * N + n] = f2bf(v);
      }
}

// ---------------------------------------------------------------------------
// RoPE cos/sin tables: cs[axis][b][l][m][g][h][{cos,sin}]
// ---------------------------------------------------------------------------
__global__ void rope_tables_kernel(const float* __restrict__ ypos,
                                   const float* __restrict__ xpos,
                                   const float* __restrict__ rope,
                                   int H, float* __restrict__ cs)
{
  int total = 2 * 2 * 64 * 4 * 2 * H;
  int idx = blockIdx.x * blockDim.x + threadIdx.x;
  if (idx >= total) return;
  int h = idx % H; int r = idx / H;
  int g = r & 1;  r >>= 1;
  int m = r & 3;  r >>= 2;
  int l = r & 63; r >>= 6;
  int b = r & 1;  int axis = r >> 1;
  const float* pos = (axis == 0 ? ypos : xpos) + (b * 64 + l) * 2;
  float fr = (H > 1) ? (float)h / (float)(H - 1) : 0.0f;
  const float PI = 3.14159265358979323846f;
  float s0 = PI / (1.0f + 29.0f * fr);
  float s1 = PI / (0.1f + 0.9f * fr);
  float f0 = rope[((m * 2 + g) * H + h) * 2 + 0] * s0;
  float f1 = rope[((m * 2 + g) * H + h) * 2 + 1] * s1;
  float phi = pos[0] * f0 + pos[1] * f1;
  cs[(size_t)idx * 2 + 0] = cosf(phi);
  cs[(size_t)idx * 2 + 1] = sinf(phi);
}

// ---------------------------------------------------------------------------
// RoPE + relayout of Q (IS_Q=1) / K (IS_Q=0):
//   dst[blk][row][192], blk over (axis,b,o,d,c,g,m) = 8192 blocks.
// ---------------------------------------------------------------------------
template<int IS_Q>
__global__ void rope_relayout_kernel(const unsigned short* __restrict__ proj_inv,
                                     const unsigned short* __restrict__ proj_fl,
                                     const float* __restrict__ cs_inv,
                                     const float* __restrict__ cs_fl,
                                     unsigned short* __restrict__ dst)
{
  long idx = (long)blockIdx.x * blockDim.x + threadIdx.x;
  if (idx >= 50331648L) return;               // 8192 * 64 * 96 pairs
  int jp = (int)(idx % 96); long r = idx / 96;
  int row = (int)(r % 64); int blk = (int)(r / 64);
  int m = blk & 3; int rest = blk >> 2;
  int g = rest & 1;  rest >>= 1;
  int cc = rest & 1; rest >>= 1;
  int d = rest & 1;  rest >>= 1;
  int o = rest & 63; rest >>= 6;
  int b = rest & 1;  int axis = rest >> 1;
  int y = (axis == 0) ? row : o;
  int x = (axis == 0) ? o : row;

  float q0, q1, c, s;
  if (jp < 32) {                                // inv stream, HI = 32
    int h = jp;
    int token = (b * 64 + y) * 64 + x;
    const unsigned short* p;
    if (IS_Q) p = proj_inv + (size_t)token * 4096 +
                  (((axis * 2 + d) * 2 + cc) * 512) + (m * 2 + g) * 64 + h * 2;
    else      p = proj_inv + (size_t)token * 1024 +
                  (((axis * 2 + d) * 2 + cc) * 128) + g * 64 + h * 2;
    q0 = bf2f(p[0]); q1 = bf2f(p[1]);
    const float* cz = cs_inv +
        ((size_t)((((axis * 2 + b) * 64 + row) * 4 + m) * 2 + g) * 32 + h) * 2;
    c = cz[0]; s = cz[1];
  } else {                                      // fl stream, HF = 16
    int jj = jp - 32; int f = jj >> 4; int h = jj & 15;
    int token4 = ((b * 64 + y) * 64 + x) * 4 + f;
    const unsigned short* p;
    if (IS_Q) p = proj_fl + (size_t)token4 * 2048 +
                  (((axis * 2 + d) * 2 + cc) * 256) + (m * 2 + g) * 32 + h * 2;
    else      p = proj_fl + (size_t)token4 * 512 +
                  (((axis * 2 + d) * 2 + cc) * 64) + g * 32 + h * 2;
    q0 = bf2f(p[0]); q1 = bf2f(p[1]);
    const float* cz = cs_fl +
        ((size_t)((((axis * 2 + b) * 64 + row) * 4 + m) * 2 + g) * 16 + h) * 2;
    c = cz[0]; s = cz[1];
  }
  float o0, o1;
  if (d == 0) { o0 = q0 * c + q1 * s; o1 = q1 * c - q0 * s; }   // r0
  else        { o0 = q0 * c - q1 * s; o1 = q1 * c + q0 * s; }   // r1
  unsigned int packed = (unsigned)f2bf(o0) | ((unsigned)f2bf(o1) << 16);
  *(unsigned int*)(dst + (size_t)blk * 12288 + row * 192 + jp * 2) = packed;
}

// ---------------------------------------------------------------------------
// V relayout, COLUMN-major per block: Va[vblk][col 192][s 64]
// ---------------------------------------------------------------------------
__global__ void v_relayout_kernel(const unsigned short* __restrict__ v_inv,
                                  const unsigned short* __restrict__ v_fl,
                                  unsigned short* __restrict__ dst)
{
  long idx = (long)blockIdx.x * blockDim.x + threadIdx.x;
  if (idx >= 25165824L) return;                 // 2048 * 64 * 192
  int j = (int)(idx % 192); long r = idx / 192;
  int row = (int)(r % 64); int vblk = (int)(r / 64);
  int g = vblk & 1; int rest = vblk >> 1;
  int cc = rest & 1; rest >>= 1;
  int d = rest & 1;  rest >>= 1;
  int o = rest & 63; rest >>= 6;
  int b = rest & 1;  int axis = rest >> 1;
  int y = (axis == 0) ? row : o;
  int x = (axis == 0) ? o : row;
  unsigned short val;
  if (j < 64) {
    int token = (b * 64 + y) * 64 + x;
    val = v_inv[(size_t)token * 1024 + (((axis * 2 + d) * 2 + cc) * 128) + g * 64 + j];
  } else {
    int jj = j - 64; int f = jj >> 5; int h = jj & 31;
    int token4 = ((b * 64 + y) * 64 + x) * 4 + f;
    val = v_fl[(size_t)token4 * 512 + (((axis * 2 + d) * 2 + cc) * 64) + g * 32 + h];
  }
  dst[(size_t)vblk * 12288 + (size_t)j * 64 + row] = val;   // transposed
}

// ---------------------------------------------------------------------------
// Attention: one block per (axis,b,o,d,c,g,m). 4 waves x 16-row strips.
//   logits[64,64] = Q[64,192] K^T ; w = mask * sigmoid(scale*logits)
//   AV[64,192]    = w V ; bf16 out.
// ---------------------------------------------------------------------------
__global__ __launch_bounds__(128) void attn_kernel(
    const unsigned short* __restrict__ Qa, const unsigned short* __restrict__ Ka,
    const unsigned short* __restrict__ Va, const unsigned char* __restrict__ mask,
    unsigned short* __restrict__ AVout)
{
  __shared__ __align__(16) unsigned short Ks[64 * 192];   // row-major [s][192]
  __shared__ __align__(16) unsigned short Vs[192 * 64];   // col-major [col][s]
  __shared__ __align__(16) unsigned short Ws[64 * 64];    // w [t][s]
  int blk = blockIdx.x;
  int rest = blk >> 5;                          // skip m, g, cc, d
  int o = rest & 63; rest >>= 6;
  int b = rest & 1;  int axis = rest >> 1;
  int vblk = blk >> 2;

  const unsigned short* kg = Ka + (size_t)blk * 12288;
  const unsigned short* vg = Va + (size_t)vblk * 12288;
  for (int i = threadIdx.x; i < 1536; i += 128) {
    async_copy_b128(Ks + i * 8, kg + i * 8);
    async_copy_b128(Vs + i * 8, vg + i * 8);
  }
  async_wait();
  __syncthreads();

  const unsigned short* Qg = Qa + (size_t)blk * 12288;
  int wave = threadIdx.x >> 5;
  int lane = threadIdx.x & 31;
  int nin = lane & 15, mh = (lane >> 4) * 8;
  const float SCALE = 0.07216878364870323f;     // 1/sqrt(192)

  // Hoist the wave's Q fragments (row strip, all 6 K-steps) out of the loop.
  AFrag qf[6];
#pragma unroll
  for (int k = 0; k < 6; ++k) load_frag(Qg, wave * 16, 192, k * 32, qf[k]);

#pragma unroll 1
  for (int sn = 0; sn < 4; ++sn) {
    v8f acc = (v8f)0.0f;
#pragma unroll
    for (int k = 0; k < 6; ++k) {
      AFrag bf;
      load_frag(Ks, sn * 16, 192, k * 32, bf);            // K^T fragment
      acc = __builtin_amdgcn_wmma_f32_16x16x32_bf16(
          false, qf[k].v, false, bf.v, (short)0, acc, false, false);
    }
    int s = sn * 16 + nin;
    int mb = (axis == 0) ? mask[b * 4096 + s * 64 + o]
                         : mask[b * 4096 + o * 64 + s];
#pragma unroll
    for (int r = 0; r < 8; ++r) {
      int t = wave * 16 + mh + r;
      float w = mb ? (1.0f / (1.0f + expf(-SCALE * acc[r]))) : 0.0f;
      Ws[t * 64 + s] = f2bf(w);
    }
  }
  __syncthreads();

  AFrag wf[2];
#pragma unroll
  for (int k = 0; k < 2; ++k) load_frag(Ws, wave * 16, 64, k * 32, wf[k]);

  unsigned short* outp = AVout + (size_t)blk * 12288;
#pragma unroll 1
  for (int nt = 0; nt < 12; ++nt) {
    v8f acc = (v8f)0.0f;
#pragma unroll
    for (int k = 0; k < 2; ++k) {
      AFrag bf;
      load_frag(Vs, nt * 16, 64, k * 32, bf);             // V^T fragment
      acc = __builtin_amdgcn_wmma_f32_16x16x32_bf16(
          false, wf[k].v, false, bf.v, (short)0, acc, false, false);
    }
#pragma unroll
    for (int r = 0; r < 8; ++r) {
      int t = wave * 16 + mh + r;
      outp[(size_t)t * 192 + nt * 16 + nin] = f2bf(acc[r]);
    }
  }
}

// ---------------------------------------------------------------------------
// Output projections: gather-A WMMA GEMM writing f32 into d_out.
// MODE 0: inv  [8192, 4096] x Wo^T -> out cols [0,512)
// MODE 1: fl   [32768,2048] x Wo^T -> out cols [512+f*256, ...)
// ---------------------------------------------------------------------------
template<int MODE>
__device__ __forceinline__ size_t av_gather(int row, int kc) {
  if (MODE == 0) {
    int b = row >> 12, y = (row >> 6) & 63, x = row & 63;
    int a2 = kc >> 10, axis = a2 >> 1, d = a2 & 1;
    int rem = kc & 1023, cc = rem >> 9, rem2 = rem & 511;
    int m = rem2 >> 7, g = (rem2 >> 6) & 1, h = rem2 & 63;
    int o = axis ? y : x, t = axis ? x : y;
    size_t blk = (((((size_t)(axis * 2 + b) * 64 + o) * 2 + d) * 2 + cc) * 2 + g) * 4 + m;
    return blk * 12288 + (size_t)t * 192 + h;
  } else {
    int b = row >> 14, y = (row >> 8) & 63, x = (row >> 2) & 63, f = row & 3;
    int a2 = kc >> 9, axis = a2 >> 1, d = a2 & 1;
    int rem = kc & 511, cc = rem >> 8, rem2 = rem & 255;
    int m = rem2 >> 6, g = (rem2 >> 5) & 1, h = rem2 & 31;
    int o = axis ? y : x, t = axis ? x : y;
    size_t blk = (((((size_t)(axis * 2 + b) * 64 + o) * 2 + d) * 2 + cc) * 2 + g) * 4 + m;
    return blk * 12288 + (size_t)t * 192 + 64 + f * 32 + h;
  }
}

template<int MODE>
__global__ __launch_bounds__(256) void out_proj_kernel(
    const unsigned short* __restrict__ AV, const unsigned short* __restrict__ Wt,
    float* __restrict__ out)
{
  constexpr int N = MODE ? 256 : 512;
  constexpr int K = MODE ? 2048 : 4096;
  __shared__ __align__(16) unsigned short As[128 * 32];
  __shared__ __align__(16) unsigned short Bs[64 * 32];
  int bm = blockIdx.y * 128, bn = blockIdx.x * 64;
  int wave = threadIdx.x >> 5;
  int wm = (wave >> 1) * 32, wn = (wave & 1) * 32;
  v8f acc[2][2];
#pragma unroll
  for (int i = 0; i < 2; ++i)
#pragma unroll
    for (int j = 0; j < 2; ++j) acc[i][j] = (v8f)0.0f;

  int brow = threadIdx.x >> 2, bcol = (threadIdx.x & 3) * 8;
  for (int k0 = 0; k0 < K; k0 += 32) {
    async_copy_b128(&Bs[brow * 32 + bcol],
                    Wt + (size_t)(bn + brow) * K + k0 + bcol);
#pragma unroll
    for (int e = 0; e < 16; ++e) {
      int idx = e * 256 + threadIdx.x;
      int row = idx >> 5, kk = idx & 31;
      As[row * 32 + kk] = AV[av_gather<MODE>(bm + row, k0 + kk)];
    }
    async_wait();
    __syncthreads();
    AFrag af, bf;
#pragma unroll
    for (int i = 0; i < 2; ++i) {
      load_frag(As, wm + 16 * i, 32, 0, af);
#pragma unroll
      for (int j = 0; j < 2; ++j) {
        load_frag(Bs, wn + 16 * j, 32, 0, bf);
        acc[i][j] = __builtin_amdgcn_wmma_f32_16x16x32_bf16(
            false, af.v, false, bf.v, (short)0, acc[i][j], false, false);
      }
    }
    __syncthreads();
  }
  int lane = threadIdx.x & 31;
  int nin = lane & 15, mh = (lane >> 4) * 8;
#pragma unroll
  for (int i = 0; i < 2; ++i)
#pragma unroll
    for (int j = 0; j < 2; ++j)
#pragma unroll
      for (int r = 0; r < 8; ++r) {
        int mr = bm + wm + 16 * i + mh + r;
        int n = bn + wn + 16 * j + nin;
        float v = acc[i][j][r];
        if (MODE == 0) {
          out[(size_t)mr * 1536 + n] = v;
        } else {
          int token = mr >> 2, f = mr & 3;
          out[(size_t)token * 1536 + 512 + f * 256 + n] = v;
        }
      }
}

// ---------------------------------------------------------------------------
// Host orchestration
// ---------------------------------------------------------------------------
extern "C" void kernel_launch(void* const* d_in, const int* in_sizes, int n_in,
                              void* d_out, int out_size, void* d_ws, size_t ws_size,
                              hipStream_t stream)
{
  (void)in_sizes; (void)n_in; (void)out_size; (void)ws_size;
  const float* x_inv   = (const float*)d_in[0];
  const float* x_fl    = (const float*)d_in[1];
  const float* ypos    = (const float*)d_in[2];
  const float* xpos    = (const float*)d_in[3];
  const unsigned char* mask = (const unsigned char*)d_in[4];
  const float* Wq_inv  = (const float*)d_in[5];
  const float* Wq_fl   = (const float*)d_in[6];
  const float* Wk_inv  = (const float*)d_in[7];
  const float* Wk_fl   = (const float*)d_in[8];
  const float* Wv_inv  = (const float*)d_in[9];
  const float* Wv_fl   = (const float*)d_in[10];
  const float* bv_inv  = (const float*)d_in[11];
  const float* bv_fl   = (const float*)d_in[12];
  const float* Wo_inv  = (const float*)d_in[13];
  const float* Wo_fl   = (const float*)d_in[14];
  const float* rope_inv = (const float*)d_in[15];
  const float* rope_fl  = (const float*)d_in[16];
  float* out = (float*)d_out;

  char* base = (char*)d_ws;
  size_t off = 0;
  auto alloc = [&](size_t bytes) -> void* {
    void* p = base + off;
    off = (off + bytes + 255) & ~(size_t)255;
    return p;
  };

  unsigned short* xinv_bf  = (unsigned short*)alloc(4194304ull * 2);
  unsigned short* xfl_bf   = (unsigned short*)alloc(8388608ull * 2);
  unsigned short* wqi_bf   = (unsigned short*)alloc(2097152ull * 2);  // transposed
  unsigned short* wki_bf   = (unsigned short*)alloc(524288ull * 2);
  unsigned short* wvi_bf   = (unsigned short*)alloc(524288ull * 2);
  unsigned short* wqf_bf   = (unsigned short*)alloc(524288ull * 2);
  unsigned short* wkf_bf   = (unsigned short*)alloc(131072ull * 2);
  unsigned short* wvf_bf   = (unsigned short*)alloc(131072ull * 2);
  unsigned short* woi_bf   = (unsigned short*)alloc(2097152ull * 2);
  unsigned short* wof_bf   = (unsigned short*)alloc(524288ull * 2);
  unsigned short* q_inv_bf = (unsigned short*)alloc(8192ull * 4096 * 2);
  unsigned short* k_inv_bf = (unsigned short*)alloc(8192ull * 1024 * 2);
  unsigned short* v_inv_bf = (unsigned short*)alloc(8192ull * 1024 * 2);
  unsigned short* q_fl_bf  = (unsigned short*)alloc(32768ull * 2048 * 2);
  unsigned short* k_fl_bf  = (unsigned short*)alloc(32768ull * 512 * 2);
  unsigned short* v_fl_bf  = (unsigned short*)alloc(32768ull * 512 * 2);
  float*          cs_inv   = (float*)alloc(65536ull * 2 * 4);
  float*          cs_fl    = (float*)alloc(32768ull * 2 * 4);
  unsigned short* Qa       = (unsigned short*)alloc(8192ull * 12288 * 2);
  unsigned short* Ka       = (unsigned short*)alloc(8192ull * 12288 * 2);
  unsigned short* Va       = (unsigned short*)alloc(2048ull * 12288 * 2);
  unsigned short* AVo      = (unsigned short*)alloc(8192ull * 12288 * 2);

  auto cvt = [&](const float* s, unsigned short* d, long n) {
    cvt_kernel<<<dim3((unsigned)((n / 8 + 255) / 256)), dim3(256), 0, stream>>>(s, d, n);
  };
  cvt(x_inv, xinv_bf, 4194304);
  cvt(x_fl,  xfl_bf,  8388608);

  auto cvtT = [&](const float* s, unsigned short* d, int K, int N) {
    long n = (long)K * N;
    cvt_t_kernel<<<dim3((unsigned)((n + 255) / 256)), dim3(256), 0, stream>>>(s, d, K, N);
  };
  cvtT(Wq_inv, wqi_bf, 512, 4096);
  cvtT(Wk_inv, wki_bf, 512, 1024);
  cvtT(Wv_inv, wvi_bf, 512, 1024);
  cvtT(Wq_fl,  wqf_bf, 256, 2048);
  cvtT(Wk_fl,  wkf_bf, 256, 512);
  cvtT(Wv_fl,  wvf_bf, 256, 512);
  cvtT(Wo_inv, woi_bf, 4096, 512);
  cvtT(Wo_fl,  wof_bf, 2048, 256);

  auto gemm = [&](const unsigned short* A, const unsigned short* Bt, unsigned short* C,
                  const float* bias, int M, int N, int K) {
    gemm_bf16_kernel<<<dim3(N / 64, M / 128), dim3(256), 0, stream>>>(A, Bt, C, bias, M, N, K);
  };
  gemm(xinv_bf, wqi_bf, q_inv_bf, nullptr, 8192, 4096, 512);
  gemm(xinv_bf, wki_bf, k_inv_bf, nullptr, 8192, 1024, 512);
  gemm(xinv_bf, wvi_bf, v_inv_bf, bv_inv,  8192, 1024, 512);
  gemm(xfl_bf,  wqf_bf, q_fl_bf,  nullptr, 32768, 2048, 256);
  gemm(xfl_bf,  wkf_bf, k_fl_bf,  nullptr, 32768, 512, 256);
  gemm(xfl_bf,  wvf_bf, v_fl_bf,  bv_fl,   32768, 512, 256);

  rope_tables_kernel<<<dim3(256), dim3(256), 0, stream>>>(ypos, xpos, rope_inv, 32, cs_inv);
  rope_tables_kernel<<<dim3(128), dim3(256), 0, stream>>>(ypos, xpos, rope_fl, 16, cs_fl);

  rope_relayout_kernel<1><<<dim3(196608), dim3(256), 0, stream>>>(
      q_inv_bf, q_fl_bf, cs_inv, cs_fl, Qa);
  rope_relayout_kernel<0><<<dim3(196608), dim3(256), 0, stream>>>(
      k_inv_bf, k_fl_bf, cs_inv, cs_fl, Ka);
  v_relayout_kernel<<<dim3(98304), dim3(256), 0, stream>>>(v_inv_bf, v_fl_bf, Va);

  attn_kernel<<<dim3(8192), dim3(128), 0, stream>>>(Qa, Ka, Va, mask, AVo);

  out_proj_kernel<0><<<dim3(8, 64), dim3(256), 0, stream>>>(AVo, woi_bf, out);
  out_proj_kernel<1><<<dim3(4, 256), dim3(256), 0, stream>>>(AVo, wof_bf, out);
}
